// GroupedQueryAttention_2293512536565
// MI455X (gfx1250) — matrix operand order repaired and checked by reading, not compiled, not run on hardware
//
#include <hip/hip_runtime.h>

// Problem constants (match reference)
#define CB   2      // batch
#define CS   2048   // seq len
#define CH   32     // query heads
#define CHKV 8      // kv heads
#define CG   4      // H / HKV
#define CD   64     // head dim
#define CHID 2048   // hidden
#define CSCALE 0.125f  // 1/sqrt(64)

typedef __attribute__((ext_vector_type(16))) __bf16 v16bf;
typedef __attribute__((ext_vector_type(8)))  __bf16 v8bf;
typedef __attribute__((ext_vector_type(8)))  float  v8f;

union BF16x16 { v16bf v; v8bf h[2]; };

static __device__ __forceinline__ v8f wmma_bf16(v16bf a, v16bf b, v8f c) {
  // (neg_a, A, neg_b, B, c_mod, C, reuse_a, reuse_b)
  return __builtin_amdgcn_wmma_f32_16x16x32_bf16(false, a, false, b, (short)0, c, false, false);
}

// ---------------------------------------------------------------- cast fp32 -> bf16
__global__ __launch_bounds__(256) void cast_bf16_kernel(const float* __restrict__ x,
                                                        __bf16* __restrict__ y, int n) {
  int i = blockIdx.x * 256 + threadIdx.x;
  if (i < n) y[i] = (__bf16)x[i];
}

// ----------------------------------------------- transpose [K,N] fp32 -> [N,K] bf16
__global__ __launch_bounds__(256) void transpose_cast_kernel(const float* __restrict__ W,
                                                             __bf16* __restrict__ Wt,
                                                             int Kdim, int Ndim) {
  __shared__ __bf16 t[32][33];
  const int nb = blockIdx.x * 32;
  const int kb = blockIdx.y * 32;
  const int tx = threadIdx.x & 31;
  const int ty = threadIdx.x >> 5;  // 0..7
  #pragma unroll
  for (int i = 0; i < 32; i += 8)
    t[ty + i][tx] = (__bf16)W[(size_t)(kb + ty + i) * Ndim + nb + tx];
  __syncthreads();
  #pragma unroll
  for (int i = 0; i < 32; i += 8)
    Wt[(size_t)(nb + ty + i) * Kdim + kb + tx] = t[tx][ty + i];
}

// ------------------------------------------------------------------- RoPE (in-place)
// x: [BH, S, D] bf16. out = x*cos + rotate_half(x)*sin, then * scale.
__global__ __launch_bounds__(256) void rope_kernel(__bf16* __restrict__ x,
                                                   const float* __restrict__ cosT,
                                                   const float* __restrict__ sinT,
                                                   int total_half, float scale) {
  int idx = blockIdx.x * 256 + threadIdx.x;
  if (idx >= total_half) return;
  int dh   = idx & 31;               // 0..31 (pair index)
  int rest = idx >> 5;
  int s    = rest & (CS - 1);
  int bh   = rest >> 11;             // CS = 2^11
  __bf16* p = x + (((size_t)bh * CS) + s) * CD;
  float x1 = (float)p[dh];
  float x2 = (float)p[dh + 32];
  float c0 = cosT[s * CD + dh];
  float s0 = sinT[s * CD + dh];
  float c1 = cosT[s * CD + dh + 32];
  float s1 = sinT[s * CD + dh + 32];
  p[dh]      = (__bf16)((x1 * c0 - x2 * s0) * scale);
  p[dh + 32] = (__bf16)((x2 * c1 + x1 * s1) * scale);
}

// ----------------------------------------------------------------- WMMA GEMM
// C = A @ Bt^T.  A: [M,K] bf16 row-major.  Bt: [N,K] bf16 row-major (pre-transposed B).
// Block: 256 threads = 8 waves, tiled 2(M) x 4(N); wave tile 16x64.
// MODE 0: fp32 store [M,N]  (final projection)
// MODE 1: Q epilogue  -> q [B,H,S,D]   bf16
// MODE 2: K epilogue  -> k [B,HKV,S,D] bf16
// MODE 3: V epilogue  -> vT [B,HKV,D,S] bf16 (transposed for PV B-fragments)
template <int MODE>
__global__ __launch_bounds__(256) void wmma_gemm_kernel(const __bf16* __restrict__ A,
                                                        const __bf16* __restrict__ Bt,
                                                        void* __restrict__ Cout,
                                                        int M, int N, int K) {
  const int lane = threadIdx.x & 31;
  const int wid  = threadIdx.x >> 5;
  const int ll = lane & 15;
  const int lh = lane >> 4;
  const int wm = wid & 1;
  const int wn = wid >> 1;
  const int m0 = blockIdx.x * 32 + wm * 16;
  const int n0 = blockIdx.y * 256 + wn * 64;

  v8f acc[4] = {v8f{0}, v8f{0}, v8f{0}, v8f{0}};
  const __bf16* arow = A + (size_t)(m0 + ll) * K;

  for (int kk = 0; kk < K; kk += 32) {
    // A fragment: lane ll holds row m0+ll; half-group lh picks K sub-range
    BF16x16 af;
    af.h[0] = *(const v8bf*)(arow + kk + lh * 16);       // K = kk + lh*16 + 0..7
    af.h[1] = *(const v8bf*)(arow + kk + lh * 16 + 8);   // K = kk + lh*16 + 8..15
    #pragma unroll
    for (int j = 0; j < 4; ++j) {
      // B fragment: lane = column n, lanes 0-15 hold K kk..kk+15, lanes 16-31 K kk+16..kk+31
      const __bf16* bcol = Bt + (size_t)(n0 + j * 16 + ll) * K + kk + lh * 16;
      v16bf bf = *(const v16bf*)bcol;
      acc[j] = wmma_bf16(af.v, bf, acc[j]);
    }
  }

  #pragma unroll
  for (int j = 0; j < 4; ++j) {
    #pragma unroll
    for (int r = 0; r < 8; ++r) {
      const int row = m0 + r + 8 * lh;   // C layout: vgpr r, half-group lh
      const int col = n0 + j * 16 + ll;
      const float val = acc[j][r];
      if (MODE == 0) {
        ((float*)Cout)[(size_t)row * N + col] = val;
      } else if (MODE == 1) {
        const int b = row >> 11, s = row & (CS - 1);
        const int hh = col >> 6, d = col & 63;
        ((__bf16*)Cout)[((((size_t)b * CH + hh) * CS) + s) * CD + d] = (__bf16)val;
      } else if (MODE == 2) {
        const int b = row >> 11, s = row & (CS - 1);
        const int hh = col >> 6, d = col & 63;
        ((__bf16*)Cout)[((((size_t)b * CHKV + hh) * CS) + s) * CD + d] = (__bf16)val;
      } else {
        const int b = row >> 11, s = row & (CS - 1);
        const int hh = col >> 6, d = col & 63;
        ((__bf16*)Cout)[((((size_t)b * CHKV + hh) * CD) + d) * CS + s] = (__bf16)val;
      }
    }
  }
}

// ------------------------------------------------------- flash attention (per block)
// One block = one (b,h) + 128 query rows (8 waves x 16). K/V chunks (32 keys) are
// staged into LDS with gfx1250 async LDS-DMA (global_load_async_to_lds_b128,
// ASYNCcnt tracked, double-buffered) and shared by all 8 waves.
// Q:[B,H,S,D] Kb:[B,HKV,S,D] Vt:[B,HKV,D,S] (bf16), mask:[B,1,S,S] fp32, O:[B,S,H*D] bf16
__global__ __launch_bounds__(256) void flash_attn_kernel(const __bf16* __restrict__ Q,
                                                         const __bf16* __restrict__ Kb,
                                                         const __bf16* __restrict__ Vt,
                                                         const float* __restrict__ mask,
                                                         __bf16* __restrict__ O) {
  __shared__ __align__(16) __bf16 ktile[2][32 * 64];  // [buf][key][d]   4KB each
  __shared__ __align__(16) __bf16 vtile[2][64 * 32];  // [buf][d][key]   4KB each
  __shared__ __align__(16) __bf16 pbuf[8][16 * 32];   // per-wave P re-layout scratch

  const int tid  = threadIdx.x;
  const int lane = tid & 31;
  const int wid  = tid >> 5;
  const int ll = lane & 15;
  const int lh = lane >> 4;

  // block task: (b, h, 128-query block); grid = B*H*(S/128) = 1024
  const int blk = blockIdx.x;
  const int b   = blk >> 9;                 // H*(S/128) = 512
  const int h   = (blk >> 4) & (CH - 1);
  const int qBase = (blk & 15) * 128 + wid * 16;
  const int hk  = h >> 2;                   // G=4

  const __bf16* kbase = Kb + (((size_t)b * CHKV + hk) * CS) * CD;
  const __bf16* vbase = Vt + (((size_t)b * CHKV + hk) * CD) * CS;
  const float*  mbase = mask + (size_t)b * CS * CS;
  __bf16* pl = pbuf[wid];

  // cooperative async-copy assignments: 256 threads move 4KB K + 4KB V per chunk
  const int krow = tid >> 3, kseg = tid & 7;   // K: 32 rows x 8 x (8 bf16)
  const int vrow = tid >> 2, vseg = tid & 3;   // V: 64 rows x 4 x (8 bf16)

  auto issue_chunk = [&](int kc, int buf) {
    const __bf16* gk = kbase + (size_t)(kc + krow) * CD + kseg * 8;
    uint32_t lk = (uint32_t)(uintptr_t)&ktile[buf][krow * 64 + kseg * 8];
    asm volatile("global_load_async_to_lds_b128 %0, %1, off" :: "v"(lk), "v"(gk) : "memory");
    const __bf16* gv = vbase + (size_t)vrow * CS + kc + vseg * 8;
    uint32_t lv = (uint32_t)(uintptr_t)&vtile[buf][vrow * 32 + vseg * 8];
    asm volatile("global_load_async_to_lds_b128 %0, %1, off" :: "v"(lv), "v"(gv) : "memory");
  };

  // Q fragments (row = qBase + ll), two K-slices of D=64
  const __bf16* qrow = Q + ((((size_t)b * CH + h) * CS) + qBase + ll) * CD;
  BF16x16 qa[2];
  #pragma unroll
  for (int kd = 0; kd < 2; ++kd) {
    const int kb = kd * 32 + lh * 16;
    qa[kd].h[0] = *(const v8bf*)(qrow + kb);
    qa[kd].h[1] = *(const v8bf*)(qrow + kb + 8);
  }

  v8f o[4] = {v8f{0}, v8f{0}, v8f{0}, v8f{0}};
  float mrun[8], lrun[8];
  #pragma unroll
  for (int r = 0; r < 8; ++r) { mrun[r] = -1e30f; lrun[r] = 0.0f; }

  issue_chunk(0, 0);

  for (int it = 0; it < CS / 32; ++it) {
    const int kc  = it * 32;
    const int buf = it & 1;
    if (it + 1 < CS / 32) {
      issue_chunk(kc + 32, buf ^ 1);
      asm volatile("s_wait_asynccnt 0x2" ::: "memory");  // chunk `it` landed in LDS
    } else {
      asm volatile("s_wait_asynccnt 0x0" ::: "memory");
    }
    __syncthreads();  // all waves' portions of chunk `it` visible

    // ---- S = Q K^T (16 queries x 32 keys) from LDS K tile
    v8f sc[2] = {v8f{0}, v8f{0}};
    #pragma unroll
    for (int j = 0; j < 2; ++j) {
      const __bf16* krowp = &ktile[buf][(j * 16 + ll) * 64 + lh * 16];
      #pragma unroll
      for (int kd = 0; kd < 2; ++kd) {
        v16bf kf = *(const v16bf*)(krowp + kd * 32);
        sc[j] = wmma_bf16(qa[kd].v, kf, sc[j]);
      }
    }
    // ---- add attention mask (softmax scale already folded into Q)
    #pragma unroll
    for (int r = 0; r < 8; ++r) {
      const int row = qBase + r + 8 * lh;
      sc[0][r] += mbase[(size_t)row * CS + kc + ll];
      sc[1][r] += mbase[(size_t)row * CS + kc + 16 + ll];
    }
    // ---- online softmax (row stats live in 16-lane half-groups)
    #pragma unroll
    for (int r = 0; r < 8; ++r) {
      float mx = fmaxf(sc[0][r], sc[1][r]);
      #pragma unroll
      for (int off = 1; off < 16; off <<= 1) mx = fmaxf(mx, __shfl_xor(mx, off, 32));
      const float nm = fmaxf(mrun[r], mx);
      const float corr = __expf(mrun[r] - nm);
      const float p0 = __expf(sc[0][r] - nm);
      const float p1 = __expf(sc[1][r] - nm);
      float rs = p0 + p1;
      #pragma unroll
      for (int off = 1; off < 16; off <<= 1) rs += __shfl_xor(rs, off, 32);
      lrun[r] = lrun[r] * corr + rs;
      mrun[r] = nm;
      #pragma unroll
      for (int j = 0; j < 4; ++j) o[j][r] *= corr;
      sc[0][r] = p0;
      sc[1][r] = p1;
    }
    // ---- re-layout P: C-layout -> A-fragment layout via wave-private LDS
    #pragma unroll
    for (int r = 0; r < 8; ++r) {
      const int row = r + 8 * lh;
      pl[row * 32 + ll]      = (__bf16)sc[0][r];
      pl[row * 32 + 16 + ll] = (__bf16)sc[1][r];
    }
    BF16x16 pa;
    pa.h[0] = *(const v8bf*)(pl + ll * 32 + lh * 16);
    pa.h[1] = *(const v8bf*)(pl + ll * 32 + lh * 16 + 8);
    // ---- O += P V from LDS V tile (4 d-tiles of 16)
    #pragma unroll
    for (int nd = 0; nd < 4; ++nd) {
      v16bf vf = *(const v16bf*)&vtile[buf][(nd * 16 + ll) * 32 + lh * 16];
      o[nd] = wmma_bf16(pa.v, vf, o[nd]);
    }

    __syncthreads();  // all waves done reading buf before it is refilled (it+2)
  }

  // ---- epilogue: O / l, write to attn [B, S, H*D]
  #pragma unroll
  for (int nd = 0; nd < 4; ++nd) {
    #pragma unroll
    for (int r = 0; r < 8; ++r) {
      const int row = qBase + r + 8 * lh;
      const int d = nd * 16 + ll;
      const float val = o[nd][r] / lrun[r];
      O[(((size_t)b * CS + row) * (CH * CD)) + h * CD + d] = (__bf16)val;
    }
  }
}

// ---------------------------------------------------------------------- launcher
extern "C" void kernel_launch(void* const* d_in, const int* in_sizes, int n_in,
                              void* d_out, int out_size, void* d_ws, size_t ws_size,
                              hipStream_t stream) {
  const float* hidden = (const float*)d_in[0];
  const float* mask   = (const float*)d_in[1];
  // d_in[2] = position_ids (arange; positions are implicit)
  const float* cosT = (const float*)d_in[3];
  const float* sinT = (const float*)d_in[4];
  const float* Wq = (const float*)d_in[5];
  const float* Wk = (const float*)d_in[6];
  const float* Wv = (const float*)d_in[7];
  const float* Wo = (const float*)d_in[8];
  float* out = (float*)d_out;

  char* ws = (char*)d_ws;
  size_t off = 0;
  auto alloc = [&](size_t bytes) -> void* {
    void* p = ws + off;
    off += (bytes + 255) & ~(size_t)255;
    return p;
  };

  const int M = CB * CS;  // 4096
  __bf16* hidb = (__bf16*)alloc((size_t)M * CHID * 2);
  __bf16* WqT  = (__bf16*)alloc((size_t)CH * CD * CHID * 2);
  __bf16* WkT  = (__bf16*)alloc((size_t)CHKV * CD * CHID * 2);
  __bf16* WvT  = (__bf16*)alloc((size_t)CHKV * CD * CHID * 2);
  __bf16* WoT  = (__bf16*)alloc((size_t)CHID * CH * CD * 2);
  __bf16* qb   = (__bf16*)alloc((size_t)CB * CH * CS * CD * 2);
  __bf16* kb   = (__bf16*)alloc((size_t)CB * CHKV * CS * CD * 2);
  __bf16* vtb  = (__bf16*)alloc((size_t)CB * CHKV * CD * CS * 2);
  __bf16* attn = (__bf16*)alloc((size_t)M * CH * CD * 2);

  // 1) convert activations + weights to bf16 (weights transposed to [N,K])
  const int nh = M * CHID;
  cast_bf16_kernel<<<nh / 256, 256, 0, stream>>>(hidden, hidb, nh);
  transpose_cast_kernel<<<dim3(CH * CD / 32, CHID / 32), 256, 0, stream>>>(Wq, WqT, CHID, CH * CD);
  transpose_cast_kernel<<<dim3(CHKV * CD / 32, CHID / 32), 256, 0, stream>>>(Wk, WkT, CHID, CHKV * CD);
  transpose_cast_kernel<<<dim3(CHKV * CD / 32, CHID / 32), 256, 0, stream>>>(Wv, WvT, CHID, CHKV * CD);
  transpose_cast_kernel<<<dim3(CHID / 32, CH * CD / 32), 256, 0, stream>>>(Wo, WoT, CH * CD, CHID);

  // 2) QKV projections (WMMA bf16)
  wmma_gemm_kernel<1><<<dim3(M / 32, (CH * CD) / 256), 256, 0, stream>>>(hidb, WqT, qb, M, CH * CD, CHID);
  wmma_gemm_kernel<2><<<dim3(M / 32, (CHKV * CD) / 256), 256, 0, stream>>>(hidb, WkT, kb, M, CHKV * CD, CHID);
  wmma_gemm_kernel<3><<<dim3(M / 32, (CHKV * CD) / 256), 256, 0, stream>>>(hidb, WvT, vtb, M, CHKV * CD, CHID);

  // 3) RoPE (softmax scale folded into Q)
  rope_kernel<<<(CB * CH * CS * 32) / 256, 256, 0, stream>>>(qb, cosT, sinT, CB * CH * CS * 32, CSCALE);
  rope_kernel<<<(CB * CHKV * CS * 32) / 256, 256, 0, stream>>>(kb, cosT, sinT, CB * CHKV * CS * 32, 1.0f);

  // 4) flash attention: one block per (b, h, 128 queries)
  flash_attn_kernel<<<CB * CH * (CS / 128), 256, 0, stream>>>(qb, kb, vtb, mask, attn);

  // 5) output projection -> fp32 d_out
  wmma_gemm_kernel<0><<<dim3(M / 32, CHID / 256), 256, 0, stream>>>(attn, WoT, out, M, CHID, CHID);

  (void)in_sizes; (void)n_in; (void)out_size; (void)ws_size;
}
